// GC_22866405884377
// MI455X (gfx1250) — matrix operand order repaired and checked by reading, not compiled, not run on hardware
//
#include <hip/hip_runtime.h>
#include <hip/hip_bf16.h>
#include <math.h>

typedef __attribute__((ext_vector_type(16))) __bf16 v16bf;
typedef __attribute__((ext_vector_type(8)))  float  v8f;
typedef __attribute__((ext_vector_type(4)))  unsigned int v4u;
typedef __attribute__((ext_vector_type(8)))  int v8i_t;
typedef __attribute__((ext_vector_type(4)))  int v4i_t;

#define NROWS 100000
#define FIN   256
#define FOUT  256
#define LDS_ROW_BYTES 528   // 256 bf16 (512B) + 16B TDM pad per row

// ---------- W [FIN][FOUT] f32  ->  Wt [FOUT][FIN] bf16 (transposed, B-friendly) ----------
__global__ void k_convW(const float* __restrict__ W, __bf16* __restrict__ Wt) {
    int tid = blockIdx.x * blockDim.x + threadIdx.x;   // 0..65535
    int c = tid >> 8;     // output column
    int k = tid & 255;    // input feature
    Wt[c * FIN + k] = (__bf16)W[k * FOUT + c];
}

// ---------- degree: init with self-loop value 1.0 ----------
__global__ void k_deg_init(float* __restrict__ dis, int n) {
    int i = blockIdx.x * blockDim.x + threadIdx.x;
    if (i < n) dis[i] = 1.0f;
}

// ---------- degree: scatter-add edge values over columns ----------
__global__ void k_deg_acc(float* __restrict__ dis, const float* __restrict__ ev,
                          const int* __restrict__ ec, int e) {
    int i = blockIdx.x * blockDim.x + threadIdx.x;
    if (i < e) unsafeAtomicAdd(&dis[ec[i]], ev[i]);
}

// ---------- dis = deg^-1/2 ----------
__global__ void k_deg_fin(float* __restrict__ dis, int n) {
    int i = blockIdx.x * blockDim.x + threadIdx.x;
    if (i < n) dis[i] = rsqrtf(dis[i]);
}

// ---------- GEMM + bias + pre-scale: xs = (x@W + b) * dis[row]; out(init) = xs ----------
// Whole Wt (256x256 bf16 = 128KB) is DMAed into LDS once per block by the
// Tensor Data Mover (with 16B/row padding for bank de-phasing), then each of
// the 8 waves computes a 16-row x 256-col tile: 16 f32 accumulator tiles,
// K-loop of 8 steps, B fragments fed by ds_load_b128.
__global__ __launch_bounds__(256) void k_gemm(const float* __restrict__ x,
                                              const __bf16* __restrict__ Wt,
                                              const float* __restrict__ bias,
                                              const float* __restrict__ dis,
                                              float* __restrict__ xs,
                                              float* __restrict__ out) {
    __shared__ __align__(16) char WtLds[FOUT * LDS_ROW_BYTES];  // 132 KB

    const int lane   = threadIdx.x & 31;
    const int wave   = threadIdx.x >> 5;
    const int laneLo = lane & 15;
    const bool hi    = lane >= 16;
    const int rowBase = (blockIdx.x * 8 + wave) * 16;

    // ---- TDM: DMA Wt[256][256] bf16 -> LDS, padding 4 DWORDs per 128 DWORDs
    {
        unsigned long long ga = (unsigned long long)Wt;
        v4u g0;
        g0[0] = 1u;                                   // count=1, user mode
        g0[1] = 0u;                                   // lds_addr = 0 (sole LDS alloc)
        g0[2] = (unsigned)ga;                         // global_addr[31:0]
        g0[3] = (unsigned)((ga >> 32) & 0x1FFFFFFu)   // global_addr[56:32]
              | 0x80000000u;                          // type=2 ("image")
        v8i_t g1;
        g1[0] = 0x07910000;          // data_size=2B | pad_enable | interval=128DW | amount=4DW
        g1[1] = (int)(256u << 16);   // tensor_dim0 = 256 (lo16 in bits 63:48)
        g1[2] = (int)(256u << 16);   // tensor_dim0 hi=0 | tensor_dim1 = 256 (lo16)
        g1[3] = (int)(256u << 16);   // tensor_dim1 hi=0 | tile_dim0 = 256
        g1[4] = 256;                 // tile_dim1 = 256, tile_dim2 = 0
        g1[5] = 256;                 // tensor_dim0_stride = 256
        g1[6] = 0;
        g1[7] = 0;
        if (wave == 0) {
#if __clang_major__ >= 23
            __builtin_amdgcn_tensor_load_to_lds(g0, g1, (v4i_t)0, (v4i_t)0, (v8i_t)0, 0);
#else
            __builtin_amdgcn_tensor_load_to_lds(g0, g1, (v4i_t)0, (v4i_t)0, 0);
#endif
            __builtin_amdgcn_s_wait_tensorcnt(0);
        }
    }
    // The TDM writes LDS behind the compiler's back (descriptor carries the LDS
    // address as plain integer). Escape WtLds + memory clobber so the B-fragment
    // ds_loads below cannot be folded to undef.
    asm volatile("" : : "v"(WtLds) : "memory");
    __syncthreads();

    v8f acc[16] = {};

    // A row this lane reads (clamped so EXEC stays all-ones for WMMA)
    int rowL = rowBase + laneLo;
    if (rowL >= NROWS) rowL = NROWS - 1;
    const float* arow = x + (size_t)rowL * FIN;

    for (int kb = 0; kb < FIN; kb += 32) {
        // ---- A fragment (contiguous k order, matching B): lanes<16 K=kb+0..15,
        //      lanes>=16 K=kb+16..31
        const float4* ap = (const float4*)(arow + kb + (hi ? 16 : 0));
        float4 f0 = ap[0], f1 = ap[1], f2 = ap[2], f3 = ap[3];
        v16bf a;
        a[0]  = (__bf16)f0.x; a[1]  = (__bf16)f0.y; a[2]  = (__bf16)f0.z; a[3]  = (__bf16)f0.w;
        a[4]  = (__bf16)f1.x; a[5]  = (__bf16)f1.y; a[6]  = (__bf16)f1.z; a[7]  = (__bf16)f1.w;
        a[8]  = (__bf16)f2.x; a[9]  = (__bf16)f2.y; a[10] = (__bf16)f2.z; a[11] = (__bf16)f2.w;
        a[12] = (__bf16)f3.x; a[13] = (__bf16)f3.y; a[14] = (__bf16)f3.z; a[15] = (__bf16)f3.w;

        const int koff2 = (kb + (hi ? 16 : 0)) * 2;   // byte offset within LDS row
#pragma unroll
        for (int t = 0; t < 16; ++t) {
            // ---- B fragment (contiguous k order, matching A): lane n -> column t*16+laneLo,
            //      lanes<16 K=kb..kb+15, lanes>=16 K=kb+16..kb+31 (LDS, padded rows)
            union { uint4 u[2]; v16bf v; } B;
            const uint4* bp = (const uint4*)(WtLds + (size_t)(t * 16 + laneLo) * LDS_ROW_BYTES + koff2);
            B.u[0] = bp[0];
            B.u[1] = bp[1];
            acc[t] = __builtin_amdgcn_wmma_f32_16x16x32_bf16(
                false, a, false, B.v, (short)0, acc[t], false, false);
        }
    }

    // ---- epilogue: +bias, *dis[row], write xs and self-loop-initialized out
    float disv[8];
    int   rows[8];
#pragma unroll
    for (int v = 0; v < 8; ++v) {
        int r = rowBase + v + (hi ? 8 : 0);
        rows[v] = r;
        disv[v] = (r < NROWS) ? dis[r] : 0.0f;
    }
#pragma unroll
    for (int t = 0; t < 16; ++t) {
        int col = t * 16 + laneLo;
        float bb = bias[col];
#pragma unroll
        for (int v = 0; v < 8; ++v) {
            if (rows[v] < NROWS) {
                float s = (acc[t][v] + bb) * disv[v];
                size_t idx = (size_t)rows[v] * FOUT + col;
                xs[idx]  = s;
                out[idx] = s;   // self-loop contribution: agg starts at xs
            }
        }
    }
}

// ---------- SpMM: one wave32 per edge; lane owns 8 contiguous floats ----------
__global__ __launch_bounds__(256) void k_spmm(const float* __restrict__ xs,
                                              const float* __restrict__ ev,
                                              const int* __restrict__ er,
                                              const int* __restrict__ ec,
                                              float* __restrict__ out, int e) {
    int edge = (int)((blockIdx.x * blockDim.x + threadIdx.x) >> 5);
    int lane = threadIdx.x & 31;
    if (edge >= e) return;
    int r = er[edge];
    int c = ec[edge];
    float v = ev[edge];
    const float4* src = (const float4*)(xs + (size_t)c * FOUT) + lane * 2;
    float4 a0 = src[0];
    float4 a1 = src[1];
    float* dst = out + (size_t)r * FOUT + lane * 8;
    unsafeAtomicAdd(dst + 0, a0.x * v);
    unsafeAtomicAdd(dst + 1, a0.y * v);
    unsafeAtomicAdd(dst + 2, a0.z * v);
    unsafeAtomicAdd(dst + 3, a0.w * v);
    unsafeAtomicAdd(dst + 4, a1.x * v);
    unsafeAtomicAdd(dst + 5, a1.y * v);
    unsafeAtomicAdd(dst + 6, a1.z * v);
    unsafeAtomicAdd(dst + 7, a1.w * v);
}

// ---------- post-scale: out *= dis[row] ----------
__global__ void k_post(float* __restrict__ out, const float* __restrict__ dis) {
    size_t i = (size_t)blockIdx.x * blockDim.x + threadIdx.x;  // over N*64 float4s
    if (i < (size_t)NROWS * (FOUT / 4)) {
        int row = (int)(i >> 6);
        float s = dis[row];
        float4* p = (float4*)out + i;
        float4 v = *p;
        v.x *= s; v.y *= s; v.z *= s; v.w *= s;
        *p = v;
    }
}

extern "C" void kernel_launch(void* const* d_in, const int* in_sizes, int n_in,
                              void* d_out, int out_size, void* d_ws, size_t ws_size,
                              hipStream_t stream) {
    const float* x  = (const float*)d_in[0];
    const float* ev = (const float*)d_in[1];
    const float* W  = (const float*)d_in[2];
    const float* b  = (const float*)d_in[3];
    const int*   er = (const int*)d_in[4];
    const int*   ec = (const int*)d_in[5];
    const int E = in_sizes[1];

    // workspace layout
    char* ws = (char*)d_ws;
    float*  dis = (float*)ws;                                   // N floats
    size_t xsOff = (((size_t)NROWS * 4) + 255) & ~(size_t)255;  // align 256B
    float*  xs  = (float*)(ws + xsOff);                          // N*256 floats
    size_t wtOff = xsOff + (size_t)NROWS * FOUT * 4;
    __bf16* Wt  = (__bf16*)(ws + wtOff);                         // 256*256 bf16
    float* out  = (float*)d_out;

    k_convW   <<<(FIN * FOUT) / 256, 256, 0, stream>>>(W, Wt);
    k_deg_init<<<(NROWS + 255) / 256, 256, 0, stream>>>(dis, NROWS);
    k_deg_acc <<<(E + 255) / 256, 256, 0, stream>>>(dis, ev, ec, E);
    k_deg_fin <<<(NROWS + 255) / 256, 256, 0, stream>>>(dis, NROWS);
    k_gemm    <<<(NROWS + 127) / 128, 256, 0, stream>>>(x, Wt, b, dis, xs, out);
    k_spmm    <<<(E + 7) / 8, 256, 0, stream>>>(xs, ev, er, ec, out, E);
    k_post    <<<(NROWS * (FOUT / 4) + 255) / 256, 256, 0, stream>>>(out, dis);
}
